// AVAttention_18683107738010
// MI455X (gfx1250) — matrix operand order repaired and checked by reading, not compiled, not run on hardware
//
#include <hip/hip_runtime.h>

// ---------------------------------------------------------------------------
// CDNA5 (gfx1250) bf16 WMMA pipeline for the AV-attention reference.
// B=16, S=1024, T=2048, D=512, C*F=2560, MEL=1280.
// ---------------------------------------------------------------------------

typedef __attribute__((ext_vector_type(16))) __bf16    v16bf;
typedef __attribute__((ext_vector_type(8)))  __bf16    v8bf;
typedef __attribute__((ext_vector_type(8)))  float     v8f;
typedef __attribute__((ext_vector_type(8)))  unsigned  v8ui;

#define WMMA_BF16(a, b, c) \
  __builtin_amdgcn_wmma_f32_16x16x32_bf16(false, (a), false, (b), (short)0, (c), false, false)

__device__ __forceinline__ __bf16 f2bf(float x) { return (__bf16)x; }

// pack two bf16 (even k low half, odd k high half) into one 32-bit LDS word
__device__ __forceinline__ unsigned packf(float lo, float hi) {
  union { __bf16 h[2]; unsigned u; } x;
  x.h[0] = (__bf16)lo; x.h[1] = (__bf16)hi; return x.u;
}
__device__ __forceinline__ unsigned packbf(__bf16 lo, __bf16 hi) {
  union { __bf16 h[2]; unsigned u; } x;
  x.h[0] = lo; x.h[1] = hi; return x.u;
}

// A fragment (16x32 bf16): row-major LDS, stride must be a multiple of 8.
// elements 0-15 = k h..h+15 (h = 16*(lane/16)) -> two aligned 16B loads.
__device__ __forceinline__ v16bf load_a_frag(const __bf16* As, int stride, int lane) {
  const int m = lane & 15;
  const int h = (lane >> 4) * 16;
  const v8bf lo = *(const v8bf*)(As + m * stride + h);
  const v8bf hi = *(const v8bf*)(As + m * stride + h + 8);
  return __builtin_shufflevector(lo, hi, 0, 1, 2, 3, 4, 5, 6, 7,
                                 8, 9, 10, 11, 12, 13, 14, 15);
}

// B fragment (32x16 bf16) from k-pair-packed LDS words: word[kw][n] = {B[2kw][n], B[2kw+1][n]}.
__device__ __forceinline__ v16bf load_b_frag_pk(const unsigned* Bw, int stride, int c0, int lane) {
  const int n = (lane & 15) + c0;
  const int kw = (lane >> 4) * 8;
  v8ui w;
#pragma unroll
  for (int v = 0; v < 8; ++v) w[v] = Bw[(kw + v) * stride + n];
  return __builtin_bit_cast(v16bf, w);
}

// ---------------------------------------------------------------------------
// Generic batched GEMM:  out[b] = A[b](MxK) @ W(KxN) + bias
// OUT_MODE 0: bf16 row-major [B,M,N];  OUT_MODE 1: fp32 [B,64,20,M] (mel transpose, NT stores)
// 256 threads (8 waves), 64x64 tile, K-step 32, double-buffered LDS staging.
// ---------------------------------------------------------------------------
template <int OUT_MODE>
__global__ __launch_bounds__(256) void gemm_bf16_wmma(
    const float* __restrict__ A, long long a_batch, int a_rs, int a_cs,
    const float* __restrict__ W, const float* __restrict__ bias,
    void* __restrict__ outp, int M, int N, int K) {
  __shared__ __align__(16) __bf16 As[2][64][40];  // stride 40 (mult of 8): aligned b128 frags
  __shared__ unsigned Bw[2][16][68];              // k-pair packed W tile
  const int tid = threadIdx.x, lane = tid & 31, wave = tid >> 5;
  const int col0 = blockIdx.x * 64, row0 = blockIdx.y * 64, b = blockIdx.z;
  const float* Ab = A + (long long)b * a_batch;
  const int r0 = (wave & 3) * 16;
  const int c0 = (wave >> 2) * 32;
  v8f acc0 = {}, acc1 = {};

  auto stage = [&](int buf, int k0) {
#pragma unroll
    for (int i = 0; i < 8; ++i) {                 // A tile: 2048 bf16
      const int e = tid + i * 256;
      int r, k;
      if (a_cs == 1) { r = e >> 5; k = e & 31; }  // coalesce along K
      else           { k = e >> 6; r = e & 63; }  // coalesce along M (g layout)
      As[buf][r][k] = f2bf(Ab[(long long)(row0 + r) * a_rs + (long long)(k0 + k) * a_cs]);
    }
#pragma unroll
    for (int i = 0; i < 4; ++i) {                 // B tile: 1024 packed words
      const int e = tid + i * 256;
      const int kw = e >> 6, c = e & 63;
      const float lo = W[(long long)(k0 + 2 * kw) * N + col0 + c];
      const float hi = W[(long long)(k0 + 2 * kw + 1) * N + col0 + c];
      Bw[buf][kw][c] = packf(lo, hi);
    }
  };

  stage(0, 0);
  int buf = 0;
  for (int k0 = 0; k0 < K; k0 += 32) {
    __syncthreads();                              // staged tile ready; prev reads done
    if (k0 + 32 < K) stage(buf ^ 1, k0 + 32);     // overlap next stage with compute
    const v16bf af = load_a_frag(&As[buf][r0][0], 40, lane);
    const v16bf b0 = load_b_frag_pk(&Bw[buf][0][0], 68, c0, lane);
    const v16bf b1 = load_b_frag_pk(&Bw[buf][0][0], 68, c0 + 16, lane);
    acc0 = WMMA_BF16(af, b0, acc0);
    acc1 = WMMA_BF16(af, b1, acc1);
    buf ^= 1;
  }

  const int mb = (lane >> 4) * 8, n = lane & 15;
#pragma unroll
  for (int v = 0; v < 8; ++v) {
    const int grow = row0 + r0 + mb + v;
    const int gc0 = col0 + c0 + n, gc1 = gc0 + 16;
    const float x0 = acc0[v] + bias[gc0];
    const float x1 = acc1[v] + bias[gc1];
    if constexpr (OUT_MODE == 0) {
      __bf16* o = (__bf16*)outp;
      o[((long long)b * M + grow) * N + gc0] = f2bf(x0);
      o[((long long)b * M + grow) * N + gc1] = f2bf(x1);
    } else {
      float* o = (float*)outp;  // out[b, gc%64, gc/64, grow]; written once -> non-temporal
      __builtin_nontemporal_store(
          x0, &o[(((long long)b * 64 + (gc0 & 63)) * 20 + (gc0 >> 6)) * M + grow]);
      __builtin_nontemporal_store(
          x1, &o[(((long long)b * 64 + (gc1 & 63)) * 20 + (gc1 >> 6)) * M + grow]);
    }
  }
}

// ---------------------------------------------------------------------------
// Flash-style masked attention: value = softmax(Q Kt / sqrt(D), mask) V
// One block = (batch, 16 rows of T). Online softmax over S in 128-col chunks.
// Q tile: async global->LDS, then held entirely in registers (16 fragments).
// QKt: K fragments straight from global (32 contiguous bytes/lane), loads
// software-pipelined one WMMA ahead. P.V: V staged as s-pair-packed LDS words.
// ---------------------------------------------------------------------------
__global__ __launch_bounds__(256) void attn_wmma(
    const __bf16* __restrict__ Qb, const __bf16* __restrict__ Kb,
    const __bf16* __restrict__ Vb, const long long* __restrict__ lengths,
    float* __restrict__ value) {
  constexpr int T = 2048, S = 1024, D = 512;
  constexpr float SCALE = 0.04419417382415922f;  // 1/sqrt(512)
  constexpr int QS = 520;    // Q tile stride (bf16), mult of 8
  constexpr int VWS = 516;   // packed V stride (u32 words)
  constexpr int CS = 136;    // score chunk stride (f32)
  constexpr int PS = 136;    // P tile stride (bf16), mult of 8
  // bytes: Q 16640 | Vw 33024 | scores 8704 | P 4352  = 62720 (<64K)
  __shared__ __align__(16) unsigned char smem[16 * QS * 2 + 16 * VWS * 4 + 16 * CS * 4 + 16 * PS * 2];
  __bf16*   Qsm = (__bf16*)(smem);
  unsigned* Vw  = (unsigned*)(smem + 16 * QS * 2);
  float*    Css = (float*)   (smem + 16 * QS * 2 + 16 * VWS * 4);
  __bf16*   Psm = (__bf16*)  (smem + 16 * QS * 2 + 16 * VWS * 4 + 16 * CS * 4);
  __shared__ float red[16][17];
  __shared__ float rmax[16], rsum[16], fac[16];

  const int tid = threadIdx.x, lane = tid & 31, wave = tid >> 5;
  const int bz = blockIdx.y, t0 = blockIdx.x * 16;
  const long long qbase  = ((long long)bz * T + t0) * D;
  const long long kvbase = (long long)bz * S * D;
  const int L = (int)lengths[bz];

  // ---- Q tile via async global->LDS (16B per lane per issue, ASYNCcnt) ----
#pragma unroll
  for (int i = 0; i < 4; ++i) {
    const int c = tid + i * 256;                 // 1024 x 16B chunks
    const int r = c >> 6, cw = c & 63;
    const __bf16* gp = Qb + qbase + (long long)r * D + cw * 8;
    const unsigned lds = (unsigned)(unsigned long long)(uintptr_t)&Qsm[r * QS + cw * 8];
    asm volatile("global_load_async_to_lds_b128 %0, %1, off"
                 :: "v"(lds), "v"(gp) : "memory");
  }
#if __has_builtin(__builtin_amdgcn_s_wait_asynccnt)
  __builtin_amdgcn_s_wait_asynccnt(0);
#else
  asm volatile("s_wait_asynccnt 0" ::: "memory");
#endif
  if (tid < 16) { rmax[tid] = -3.0e38f; rsum[tid] = 0.f; }
  __syncthreads();

  // Q fragments are invariant across all S chunks: hold all 16 in registers.
  v16bf qfr[16];
#pragma unroll
  for (int j = 0; j < 16; ++j) qfr[j] = load_a_frag(Qsm + j * 32, QS, lane);

  v8f oacc[4] = {};
  const int mb = (lane >> 4) * 8, nn = lane & 15;
  const int row = tid >> 4, sub = tid & 15;

  for (int s0 = 0; s0 < S; s0 += 128) {
    // ---- scores = Q Kt for this 16x128 chunk (K direct from global) ----
    v8f sacc = {};
    const __bf16* kptr = Kb + kvbase +
        (long long)(s0 + wave * 16 + nn) * D + ((lane >> 4) * 16);
    v16bf kf = *(const v16bf*)(kptr);            // pipeline depth 1
#pragma unroll
    for (int j = 0; j < 16; ++j) {
      const v16bf cur = kf;
      if (j < 15) kf = *(const v16bf*)(kptr + (j + 1) * 32);
      sacc = WMMA_BF16(qfr[j], cur, sacc);
    }
    __syncthreads();
#pragma unroll
    for (int v = 0; v < 8; ++v) {                // scale + length mask
      const int scol = wave * 16 + nn;
      float val = sacc[v] * SCALE;
      if (s0 + scol >= L) val = -3.0e38f;
      Css[(mb + v) * CS + scol] = val;
    }
    __syncthreads();
    // ---- online softmax update ----
    float mx = -3.0e38f;
#pragma unroll
    for (int s = 0; s < 128; s += 16) mx = fmaxf(mx, Css[row * CS + s + sub]);
    red[row][sub] = mx;
    __syncthreads();
    if (tid < 16) {
      float m2 = rmax[tid];
#pragma unroll
      for (int j = 0; j < 16; ++j) m2 = fmaxf(m2, red[tid][j]);
      fac[tid] = __expf(rmax[tid] - m2);
      rmax[tid] = m2;
    }
    __syncthreads();
    const float mrow = rmax[row];
    float sm = 0.f;
#pragma unroll
    for (int s = 0; s < 128; s += 16) {
      const float e = __expf(Css[row * CS + s + sub] - mrow);
      Psm[row * PS + s + sub] = f2bf(e);
      sm += e;
    }
    red[row][sub] = sm;
    __syncthreads();
    if (tid < 16) {
      float s2 = 0.f;
#pragma unroll
      for (int j = 0; j < 16; ++j) s2 += red[tid][j];
      rsum[tid] = rsum[tid] * fac[tid] + s2;
    }
    __syncthreads();
#pragma unroll
    for (int t = 0; t < 4; ++t)                  // rescale running output
#pragma unroll
      for (int v = 0; v < 8; ++v) oacc[t][v] *= fac[mb + v];
    // ---- accumulate P @ V for this chunk ----
    for (int ss = 0; ss < 128; ss += 32) {
      __syncthreads();
#pragma unroll
      for (int i = 0; i < 32; ++i) {             // stage V as s-pair packed words
        const int e = tid + i * 256;             // 8192 words
        const int sw = e >> 9, d = e & 511;
        const long long rb = kvbase + (long long)(s0 + ss + 2 * sw) * D + d;
        Vw[sw * VWS + d] = packbf(Vb[rb], Vb[rb + D]);
      }
      __syncthreads();
      const v16bf pf = load_a_frag(Psm + ss, PS, lane);
#pragma unroll
      for (int t = 0; t < 4; ++t) {
        const v16bf vf = load_b_frag_pk(Vw, VWS, wave * 64 + t * 16, lane);
        oacc[t] = WMMA_BF16(pf, vf, oacc[t]);
      }
    }
    __syncthreads();
  }

  // normalize and store value tile (fp32, feeds mel GEMM; keep cached)
#pragma unroll
  for (int t = 0; t < 4; ++t)
#pragma unroll
    for (int v = 0; v < 8; ++v) {
      const float inv = 1.0f / rsum[mb + v];
      value[((long long)bz * T + t0 + mb + v) * D + wave * 64 + t * 16 + nn] = oacc[t][v] * inv;
    }
}

// ---------------------------------------------------------------------------
extern "C" void kernel_launch(void* const* d_in, const int* in_sizes, int n_in,
                              void* d_out, int out_size, void* d_ws, size_t ws_size,
                              hipStream_t stream) {
  (void)in_sizes; (void)n_in; (void)out_size; (void)ws_size;
  constexpr int B = 16, S = 1024, T = 2048, D = 512, CF = 2560, MEL = 1280;
  const float* ph = (const float*)d_in[0];
  const float* g  = (const float*)d_in[1];
  const long long* lengths = (const long long*)d_in[2];
  const float* Wk   = (const float*)d_in[3];  const float* bk   = (const float*)d_in[4];
  const float* Wv   = (const float*)d_in[5];  const float* bv   = (const float*)d_in[6];
  const float* Wq   = (const float*)d_in[7];  const float* bq   = (const float*)d_in[8];
  const float* Wmel = (const float*)d_in[9];  const float* bmel = (const float*)d_in[10];

  // workspace: K bf16 | V bf16 | Q bf16 | value fp32   (16+16+32+64 MB = 128 MB)
  char* ws = (char*)d_ws;
  __bf16* Kb = (__bf16*)(ws);
  __bf16* Vb = (__bf16*)(ws + (size_t)B * S * D * 2);
  __bf16* Qb = (__bf16*)(ws + (size_t)B * S * D * 4);
  float* value = (float*)(ws + (size_t)B * S * D * 4 + (size_t)B * T * D * 2);

  const dim3 blk(256);
  gemm_bf16_wmma<0><<<dim3(D / 64, S / 64, B), blk, 0, stream>>>(
      ph, (long long)S * D, D, 1, Wk, bk, Kb, S, D, D);
  gemm_bf16_wmma<0><<<dim3(D / 64, S / 64, B), blk, 0, stream>>>(
      ph, (long long)S * D, D, 1, Wv, bv, Vb, S, D, D);
  gemm_bf16_wmma<0><<<dim3(D / 64, T / 64, B), blk, 0, stream>>>(
      g, (long long)CF * T, 1, T, Wq, bq, Qb, T, D, CF);
  attn_wmma<<<dim3(T / 16, B), blk, 0, stream>>>(Qb, Kb, Vb, lengths, value);
  gemm_bf16_wmma<1><<<dim3(MEL / 64, T / 64, B), blk, 0, stream>>>(
      value, (long long)T * D, D, 1, Wmel, bmel, d_out, T, MEL, D);
}
